// GraphNeuralNetwork_20873541058681
// MI455X (gfx1250) — matrix operand order repaired and checked by reading, not compiled, not run on hardware
//
#include <hip/hip_runtime.h>
#include <hip/hip_bf16.h>
#include <math.h>

typedef __bf16 bf16;
typedef __attribute__((ext_vector_type(16))) __bf16 v16bf;
typedef __attribute__((ext_vector_type(8)))  __bf16 v8bf;
typedef __attribute__((ext_vector_type(8)))  float  v8f;

#define D_IN  128
#define D_HID 256
#define D_OUT 128
#define NGRAPH 128

// ---------------------------------------------------------------------------
// Utility kernels
// ---------------------------------------------------------------------------
__global__ void zero_f32(float* __restrict__ p, size_t n) {
  size_t i = (size_t)blockIdx.x * blockDim.x + threadIdx.x;
  if (i < n) p[i] = 0.0f;
}

__global__ void f32_to_bf16(const float* __restrict__ in, bf16* __restrict__ out, size_t n) {
  size_t i = (size_t)blockIdx.x * blockDim.x + threadIdx.x;
  if (i < n) out[i] = (bf16)in[i];
}

// W is [K, ND] row-major f32 -> Wt [ND, K] row-major bf16 (i.e. column-major W)
__global__ void convert_transpose(const float* __restrict__ W, bf16* __restrict__ Wt,
                                  int K, int ND) {
  int idx = blockIdx.x * blockDim.x + threadIdx.x;
  if (idx >= K * ND) return;
  int k = idx / ND, n = idx % ND;
  Wt[(size_t)n * K + k] = (bf16)W[idx];
}

// ---------------------------------------------------------------------------
// Degree / normalization: deg[i] = (#edges with dst==i) + 1 (self loop),
// dinv[i] = rsqrt(deg[i])
// ---------------------------------------------------------------------------
__global__ void deg_accum(const int* __restrict__ dst, float* __restrict__ deg, int E) {
  int e = blockIdx.x * blockDim.x + threadIdx.x;
  if (e < E) atomicAdd(&deg[dst[e]], 1.0f);
}

__global__ void deg_to_dinv(float* __restrict__ deg, int N) {
  int i = blockIdx.x * blockDim.x + threadIdx.x;
  if (i < N) deg[i] = rsqrtf(deg[i] + 1.0f);   // +1 for self loop; always >= 1
}

// ---------------------------------------------------------------------------
// WMMA GEMM: C[nrows, ND] = A[nrows, KD] (bf16) x W (via Bt[ND, KD] bf16)
// One wave per 16x16 output tile. nrows must be a multiple of 16 (100000 = 6250*16).
// ---------------------------------------------------------------------------
template<int KD, int ND>
__global__ __launch_bounds__(256) void gcn_gemm_bf16(
    const bf16* __restrict__ A, const bf16* __restrict__ Bt,
    float* __restrict__ C, int nrows) {
  const int tilesN = ND / 16;
  int wave = blockIdx.x * 8 + (threadIdx.x >> 5);
  int lane = threadIdx.x & 31;
  int tilesM = nrows >> 4;
  if (wave >= tilesM * tilesN) return;
  int tm = wave / tilesN;
  int tn = wave % tilesN;
  int half = lane >> 4;        // 0: lanes 0-15, 1: lanes 16-31
  int l    = lane & 15;

  const bf16* arow = A  + (size_t)(tm * 16 + l) * KD;   // A row m = l (per lane)
  const bf16* bcol = Bt + (size_t)(tn * 16 + l) * KD;   // W column n = l (per lane)

  v8f acc = {0.f, 0.f, 0.f, 0.f, 0.f, 0.f, 0.f, 0.f};
#pragma unroll
  for (int k0 = 0; k0 < KD; k0 += 32) {
    // A 16x32 bf16 fragment: lane holds K in [k0+8*half, +8) and [k0+16+8*half, +8)
    v8bf alo = *(const v8bf*)(arow + k0 + half * 8);
    v8bf ahi = *(const v8bf*)(arow + k0 + 16 + half * 8);
    // B 32x16 bf16 fragment: lane holds K in [k0+8*half, +8) and [k0+16+8*half, +8)
    v8bf blo = *(const v8bf*)(bcol + k0 + half * 8);
    v8bf bhi = *(const v8bf*)(bcol + k0 + 16 + half * 8);
    v16bf a, b;
#pragma unroll
    for (int i = 0; i < 8; ++i) {
      a[i] = alo[i]; a[8 + i] = ahi[i];
      b[i] = blo[i]; b[8 + i] = bhi[i];
    }
    acc = __builtin_amdgcn_wmma_f32_16x16x32_bf16(
        /*neg_a=*/false, a, /*neg_b=*/false, b,
        /*c_mod=*/(short)0, acc, /*reuse_a=*/false, /*reuse_b=*/false);
  }

  // D layout: VGPR r -> row (tm*16 + 8*half + r), col (tn*16 + l)
  float* crow = C + (size_t)(tm * 16 + half * 8) * ND + tn * 16 + l;
#pragma unroll
  for (int r = 0; r < 8; ++r) crow[(size_t)r * ND] = acc[r];
}

// ---------------------------------------------------------------------------
// Edge scatter: acc[dst] += T[src] * dinv[src] * dinv[dst]  (L2-resident atomics)
// blockDim = 256, 256/ND edges per block.
// ---------------------------------------------------------------------------
template<int ND>
__global__ __launch_bounds__(256) void scatter_edges(
    const float* __restrict__ T, const int* __restrict__ src,
    const int* __restrict__ dst, const float* __restrict__ dinv,
    float* __restrict__ acc, int E) {
  constexpr int EPB = 256 / ND;
  int t = threadIdx.x;
  int e = blockIdx.x * EPB + t / ND;
  if (e >= E) return;
  int f = t % ND;
  int s = src[e];
  int d = dst[e];
  float w = dinv[s] * dinv[d];
  atomicAdd(acc + (size_t)d * ND + f, T[(size_t)s * ND + f] * w);
}

// ---------------------------------------------------------------------------
// Epilogues: add self-loop term + bias, optional relu, emit bf16 for next layer
// ---------------------------------------------------------------------------
__global__ void epilogue_relu_bf16(const float* __restrict__ acc, const float* __restrict__ T,
                                   const float* __restrict__ dinv, const float* __restrict__ bias,
                                   bf16* __restrict__ out, int N, int ND) {
  size_t idx = (size_t)blockIdx.x * blockDim.x + threadIdx.x;
  if (idx >= (size_t)N * ND) return;
  int i = (int)(idx / ND);
  int f = (int)(idx % ND);
  float di = dinv[i];
  float v = acc[idx] + T[idx] * di * di + bias[f];
  out[idx] = (bf16)fmaxf(v, 0.0f);
}

// last conv: no relu, write f32 in place into acc (becomes pooled features)
__global__ void epilogue_f32_inplace(float* __restrict__ acc, const float* __restrict__ T,
                                     const float* __restrict__ dinv, const float* __restrict__ bias,
                                     int N, int ND) {
  size_t idx = (size_t)blockIdx.x * blockDim.x + threadIdx.x;
  if (idx >= (size_t)N * ND) return;
  int i = (int)(idx / ND);
  int f = (int)(idx % ND);
  float di = dinv[i];
  acc[idx] = acc[idx] + T[idx] * di * di + bias[f];
}

// ---------------------------------------------------------------------------
// Global mean pool: atomic sums per graph (batch is sorted, G=128)
// ---------------------------------------------------------------------------
__global__ __launch_bounds__(128) void pool_accum(const float* __restrict__ H,
                                                  const int* __restrict__ batch,
                                                  float* __restrict__ pool,
                                                  float* __restrict__ cnt, int N) {
  int i = blockIdx.x;
  if (i >= N) return;
  int g = batch[i];
  int f = threadIdx.x;
  atomicAdd(&pool[(size_t)g * D_OUT + f], H[(size_t)i * D_OUT + f]);
  if (f == 0) atomicAdd(&cnt[g], 1.0f);
}

// ---------------------------------------------------------------------------
// MLP head: one block per graph. hid = relu(g@Wf1 + bf1); out = sigmoid(hid@Wf2 + bf2)
// ---------------------------------------------------------------------------
__global__ __launch_bounds__(256) void mlp_head(const float* __restrict__ pool,
                                                const float* __restrict__ cnt,
                                                const float* __restrict__ Wf1,
                                                const float* __restrict__ bf1,
                                                const float* __restrict__ Wf2,
                                                const float* __restrict__ bf2,
                                                float* __restrict__ out) {
  __shared__ float g[D_OUT];
  __shared__ float red[256];
  int gi = blockIdx.x;
  int t  = threadIdx.x;
  float inv = 1.0f / fmaxf(cnt[gi], 1.0f);
  if (t < D_OUT) g[t] = pool[(size_t)gi * D_OUT + t] * inv;
  __syncthreads();
  // hid[t], t in [0,256)
  float s = bf1[t];
#pragma unroll 4
  for (int k = 0; k < D_OUT; ++k) s += g[k] * Wf1[(size_t)k * D_HID + t];
  float hid = fmaxf(s, 0.0f);
  red[t] = hid * Wf2[t];
  __syncthreads();
  for (int sft = 128; sft > 0; sft >>= 1) {
    if (t < sft) red[t] += red[t + sft];
    __syncthreads();
  }
  if (t == 0) out[gi] = 1.0f / (1.0f + expf(-(red[0] + bf2[0])));
}

// ---------------------------------------------------------------------------
// Launch
// ---------------------------------------------------------------------------
static inline int cdiv(long long a, long long b) { return (int)((a + b - 1) / b); }

extern "C" void kernel_launch(void* const* d_in, const int* in_sizes, int n_in,
                              void* d_out, int out_size, void* d_ws, size_t ws_size,
                              hipStream_t stream) {
  const int N = in_sizes[0] / D_IN;
  const int E = in_sizes[1] / 2;

  const float* x     = (const float*)d_in[0];
  const int*   ei    = (const int*)d_in[1];
  const int*   batch = (const int*)d_in[2];
  const float* W0 = (const float*)d_in[3];  const float* b0 = (const float*)d_in[4];
  const float* W1 = (const float*)d_in[5];  const float* b1 = (const float*)d_in[6];
  const float* W2 = (const float*)d_in[7];  const float* b2 = (const float*)d_in[8];
  const float* Wf1 = (const float*)d_in[9];  const float* bf1 = (const float*)d_in[10];
  const float* Wf2 = (const float*)d_in[11]; const float* bf2 = (const float*)d_in[12];

  const int* src = ei;        // edge_index[0]
  const int* dst = ei + E;    // edge_index[1]

  // Workspace carve-out (256B aligned slabs)
  char* p = (char*)d_ws;
  auto alloc = [&](size_t bytes) -> void* {
    void* r = (void*)p;
    p += (bytes + 255) & ~(size_t)255;
    return r;
  };
  float* dinv = (float*)alloc((size_t)N * 4);
  bf16*  Wt0  = (bf16*)alloc((size_t)D_IN  * D_HID * 2);
  bf16*  Wt1  = (bf16*)alloc((size_t)D_HID * D_HID * 2);
  bf16*  Wt2  = (bf16*)alloc((size_t)D_HID * D_OUT * 2);
  bf16*  Hb   = (bf16*)alloc((size_t)N * D_HID * 2);   // bf16 activations (layer input)
  float* T    = (float*)alloc((size_t)N * D_HID * 4);  // GEMM output h@W
  float* ACC  = (float*)alloc((size_t)N * D_HID * 4);  // scatter accumulator
  float* pool = (float*)alloc((size_t)NGRAPH * D_OUT * 4);
  float* cnt  = (float*)alloc((size_t)NGRAPH * 4);
  (void)ws_size; (void)n_in; (void)out_size;

  const int TB = 256;

  // 1) degree -> dinv
  zero_f32<<<cdiv(N, TB), TB, 0, stream>>>(dinv, (size_t)N);
  deg_accum<<<cdiv(E, TB), TB, 0, stream>>>(dst, dinv, E);
  deg_to_dinv<<<cdiv(N, TB), TB, 0, stream>>>(dinv, N);

  // 2) weights -> transposed bf16; x -> bf16
  convert_transpose<<<cdiv(D_IN * D_HID, TB), TB, 0, stream>>>(W0, Wt0, D_IN, D_HID);
  convert_transpose<<<cdiv(D_HID * D_HID, TB), TB, 0, stream>>>(W1, Wt1, D_HID, D_HID);
  convert_transpose<<<cdiv(D_HID * D_OUT, TB), TB, 0, stream>>>(W2, Wt2, D_HID, D_OUT);
  f32_to_bf16<<<cdiv((long long)N * D_IN, TB), TB, 0, stream>>>(x, Hb, (size_t)N * D_IN);

  const int tilesM = N / 16;  // N = 100000 = 6250 * 16

  // ---- Layer 0: [N,128] @ [128,256], relu ----
  {
    int tiles = tilesM * (D_HID / 16);
    gcn_gemm_bf16<D_IN, D_HID><<<cdiv(tiles, 8), TB, 0, stream>>>(Hb, Wt0, T, N);
    zero_f32<<<cdiv((long long)N * D_HID, TB), TB, 0, stream>>>(ACC, (size_t)N * D_HID);
    scatter_edges<D_HID><<<E, TB, 0, stream>>>(T, src, dst, dinv, ACC, E);
    epilogue_relu_bf16<<<cdiv((long long)N * D_HID, TB), TB, 0, stream>>>(
        ACC, T, dinv, b0, Hb, N, D_HID);
  }

  // ---- Layer 1: [N,256] @ [256,256], relu ----
  {
    int tiles = tilesM * (D_HID / 16);
    gcn_gemm_bf16<D_HID, D_HID><<<cdiv(tiles, 8), TB, 0, stream>>>(Hb, Wt1, T, N);
    zero_f32<<<cdiv((long long)N * D_HID, TB), TB, 0, stream>>>(ACC, (size_t)N * D_HID);
    scatter_edges<D_HID><<<E, TB, 0, stream>>>(T, src, dst, dinv, ACC, E);
    epilogue_relu_bf16<<<cdiv((long long)N * D_HID, TB), TB, 0, stream>>>(
        ACC, T, dinv, b1, Hb, N, D_HID);
  }

  // ---- Layer 2: [N,256] @ [256,128], no relu -> ACC holds f32 node features ----
  {
    int tiles = tilesM * (D_OUT / 16);
    gcn_gemm_bf16<D_HID, D_OUT><<<cdiv(tiles, 8), TB, 0, stream>>>(Hb, Wt2, T, N);
    zero_f32<<<cdiv((long long)N * D_OUT, TB), TB, 0, stream>>>(ACC, (size_t)N * D_OUT);
    scatter_edges<D_OUT><<<cdiv(E, 2), TB, 0, stream>>>(T, src, dst, dinv, ACC, E);
    epilogue_f32_inplace<<<cdiv((long long)N * D_OUT, TB), TB, 0, stream>>>(
        ACC, T, dinv, b2, N, D_OUT);
  }

  // ---- Global mean pool + MLP head ----
  zero_f32<<<cdiv(NGRAPH * D_OUT, TB), TB, 0, stream>>>(pool, (size_t)NGRAPH * D_OUT);
  zero_f32<<<1, NGRAPH, 0, stream>>>(cnt, (size_t)NGRAPH);
  pool_accum<<<N, D_OUT, 0, stream>>>(ACC, batch, pool, cnt, N);
  mlp_head<<<NGRAPH, TB, 0, stream>>>(pool, cnt, Wf1, bf1, Wf2, bf2, (float*)d_out);
}
